// Loss_52879637348366
// MI455X (gfx1250) — matrix operand order repaired and checked by reading, not compiled, not run on hardware
//
#include <hip/hip_runtime.h>
#include <math.h>

#define EMB    512
#define VOCAB  28996
#define NROWS  8192
#define VS     8                       // vocab chunks
#define BM     256                     // rows per workgroup (8 waves x 32)
#define RB     (NROWS / BM)            // 32 row blocks
#define VCH    ((VOCAB + VS - 1) / VS) // 3625 cols per chunk

typedef __attribute__((ext_vector_type(16))) __bf16 v16bf;
typedef __attribute__((ext_vector_type(8)))  __bf16 bf16x8;
typedef __attribute__((ext_vector_type(8)))  float  v8f;

// sizes (elements)
#define XN ((size_t)NROWS * EMB)   // 4,194,304
#define WN ((size_t)VOCAB * EMB)   // 14,845,952
// float scratch region (elements): pm, ps, tp (N*VS each), m, s, py (N each)
#define NFLOAT ((size_t)NROWS * VS * 3 + (size_t)NROWS * 3)
// total workspace needed for the pre-converted path (bytes)
#define WS_NEED (NFLOAT * 4 + (2 * XN + 2 * WN) * 2)

// ---- f32 -> (hi,lo) bf16 split: a ~= hi + lo ----
__device__ inline void cvt_split16(const float* __restrict__ f, v16bf& hi, v16bf& lo) {
#pragma unroll
    for (int i = 0; i < 16; ++i) {
        float v = f[i];
        __bf16 h = (__bf16)v;
        hi[i] = h;
        lo[i] = (__bf16)(v - (float)h);
    }
}

__device__ inline v16bf join8(bf16x8 a, bf16x8 b) {
    return __builtin_shufflevector(a, b, 0, 1, 2, 3, 4, 5, 6, 7,
                                   8, 9, 10, 11, 12, 13, 14, 15);
}

// A fragment (16x32 MxK): lanes 0-15 K={k0..k0+7, k0+16..23}; lanes 16-31 +8.
__device__ inline v16bf frag_a_bf16(const __bf16* __restrict__ rowp, int k0, int half) {
    const __bf16* p = rowp + k0 + half * 8;
    bf16x8 c0 = *(const bf16x8*)p;
    bf16x8 c1 = *(const bf16x8*)(p + 16);
    return join8(c0, c1);
}

// B fragment (32x16 KxN): lanes 0-15 K={k0..k0+7, k0+16..23}; lanes 16-31 +8.
__device__ inline v16bf frag_b_bf16(const __bf16* __restrict__ colp, int k0, int half) {
    const __bf16* p = colp + k0 + half * 8;
    bf16x8 c0 = *(const bf16x8*)p;
    bf16x8 c1 = *(const bf16x8*)(p + 16);
    return join8(c0, c1);
}

// f32 fallbacks (convert in-loop)
__device__ inline void frag_a_f32(const float* __restrict__ rowp, int k0, int half,
                                  v16bf& hi, v16bf& lo) {
    const float4* p = (const float4*)(rowp + k0 + half * 8);
    float4 a = p[0], b = p[1], c = p[4], d = p[5];
    float f[16] = {a.x,a.y,a.z,a.w, b.x,b.y,b.z,b.w,
                   c.x,c.y,c.z,c.w, d.x,d.y,d.z,d.w};
    cvt_split16(f, hi, lo);
}
__device__ inline void frag_b_f32(const float* __restrict__ colp, int k0, int half,
                                  v16bf& hi, v16bf& lo) {
    const float4* p = (const float4*)(colp + k0 + half * 8);
    float4 a = p[0], b = p[1], c = p[4], d = p[5];
    float f[16] = {a.x,a.y,a.z,a.w, b.x,b.y,b.z,b.w,
                   c.x,c.y,c.z,c.w, d.x,d.y,d.z,d.w};
    cvt_split16(f, hi, lo);
}

__device__ inline v8f wmma3_bf16(v16bf ah, v16bf al, v16bf bh, v16bf bl, v8f acc) {
    acc = __builtin_amdgcn_wmma_f32_16x16x32_bf16(false, ah, false, bh, (short)0, acc, false, false);
    acc = __builtin_amdgcn_wmma_f32_16x16x32_bf16(false, ah, false, bl, (short)0, acc, false, false);
    acc = __builtin_amdgcn_wmma_f32_16x16x32_bf16(false, al, false, bh, (short)0, acc, false, false);
    return acc;
}

// one-shot f32 -> (hi, lo) bf16 arrays, 8 elements / thread
__global__ __launch_bounds__(256) void split_bf16_kernel(
    const float* __restrict__ src, __bf16* __restrict__ hi, __bf16* __restrict__ lo,
    long n)
{
    long i = ((long)blockIdx.x * 256 + threadIdx.x) * 8;
    if (i >= n) return;
    const float4* p = (const float4*)(src + i);
    float4 f0 = p[0], f1 = p[1];
    float f[8] = {f0.x, f0.y, f0.z, f0.w, f1.x, f1.y, f1.z, f1.w};
    bf16x8 h, l;
#pragma unroll
    for (int e = 0; e < 8; ++e) {
        __bf16 hb = (__bf16)f[e];
        h[e] = hb;
        l[e] = (__bf16)(f[e] - (float)hb);
    }
    *(bf16x8*)(hi + i) = h;
    *(bf16x8*)(lo + i) = l;
}

// PASS 1: per-(row, chunk) online softmax stats (m, s) over logits.
// PASS 2: per-(row, chunk) partial t = sum_v exp( exp(l - m) / s ).
// PRE: operands pre-split to bf16 in workspace (pure load+WMMA inner loop).
template <int PASS, bool PRE>
__global__ __launch_bounds__(256) void gemm_softmax_kernel(
    const float* __restrict__ xf, const float* __restrict__ Wf,
    const __bf16* __restrict__ xh, const __bf16* __restrict__ xl,
    const __bf16* __restrict__ wh, const __bf16* __restrict__ wl,
    const float* __restrict__ bias,
    const float* __restrict__ mrow, const float* __restrict__ srow,  // PASS 2 only
    float* __restrict__ out0, float* __restrict__ out1)              // pm/ps or tp/-
{
    const int rb   = blockIdx.x;
    const int vs   = blockIdx.y;
    const int wave = threadIdx.x >> 5;
    const int lane = threadIdx.x & 31;
    const int half = lane >> 4;
    const int idx  = lane & 15;

    const int rbase  = rb * BM + wave * 32;      // 2 stripes of 16 rows
    const int vstart = vs * VCH;
    const int vend   = min(VOCAB, vstart + VCH);

    // state [stripe][r] maps to row rbase + s*16 + 8*half + r
    float stM[2][8], stS[2][8];
    float mloc[2][8], sinv[2][8];
#pragma unroll
    for (int s = 0; s < 2; ++s)
#pragma unroll
        for (int r = 0; r < 8; ++r) { stM[s][r] = -INFINITY; stS[s][r] = 0.0f; }
    if (PASS == 2) {
#pragma unroll
        for (int s = 0; s < 2; ++s)
#pragma unroll
            for (int r = 0; r < 8; ++r) {
                int row = rbase + s * 16 + 8 * half + r;
                mloc[s][r] = mrow[row];
                sinv[s][r] = 1.0f / srow[row];
            }
    }

    const float*  arf[2];
    const __bf16* arh[2];
    const __bf16* arl[2];
#pragma unroll
    for (int s = 0; s < 2; ++s) {
        size_t ro = (size_t)(rbase + s * 16 + idx) * EMB;
        if (PRE) { arh[s] = xh + ro; arl[s] = xl + ro; }
        else     { arf[s] = xf + ro; }
    }

    for (int n0 = vstart; n0 < vend; n0 += 64) {
        v8f acc[2][4] = {};
        size_t co[4];
#pragma unroll
        for (int j = 0; j < 4; ++j) {
            int c = min(n0 + j * 16 + idx, VOCAB - 1);
            co[j] = (size_t)c * EMB;
        }

#pragma unroll 2
        for (int k0 = 0; k0 < EMB; k0 += 32) {
            v16bf Ah[2], Al[2];
#pragma unroll
            for (int s = 0; s < 2; ++s) {
                if (PRE) {
                    Ah[s] = frag_a_bf16(arh[s], k0, half);
                    Al[s] = frag_a_bf16(arl[s], k0, half);
                } else {
                    frag_a_f32(arf[s], k0, half, Ah[s], Al[s]);
                }
            }
#pragma unroll
            for (int j = 0; j < 4; ++j) {
                v16bf Bh, Bl;
                if (PRE) {
                    Bh = frag_b_bf16(wh + co[j], k0, half);
                    Bl = frag_b_bf16(wl + co[j], k0, half);
                } else {
                    frag_b_f32(Wf + co[j], k0, half, Bh, Bl);
                }
#pragma unroll
                for (int s = 0; s < 2; ++s)
                    acc[s][j] = wmma3_bf16(Ah[s], Al[s], Bh, Bl, acc[s][j]);
            }
        }

        // consume logits
#pragma unroll
        for (int j = 0; j < 4; ++j) {
            int  c     = n0 + j * 16 + idx;
            bool valid = (c < vend);
            float bv   = bias[min(c, VOCAB - 1)];
#pragma unroll
            for (int s = 0; s < 2; ++s)
#pragma unroll
                for (int r = 0; r < 8; ++r) {
                    float l = acc[s][j][r] + bv;
                    if (PASS == 1) {
                        if (valid) {
                            float nm = fmaxf(stM[s][r], l);
                            stS[s][r] = stS[s][r] * __expf(stM[s][r] - nm) + __expf(l - nm);
                            stM[s][r] = nm;
                        }
                    } else {
                        if (valid) {
                            float p = __expf(l - mloc[s][r]) * sinv[s][r];
                            stS[s][r] += __expf(p);
                        }
                    }
                }
        }
    }

    // cross-lane reduction within each 16-lane half (xor keeps bit4 fixed)
    if (PASS == 1) {
#pragma unroll
        for (int off = 1; off < 16; off <<= 1) {
#pragma unroll
            for (int s = 0; s < 2; ++s)
#pragma unroll
                for (int r = 0; r < 8; ++r) {
                    float m2 = __shfl_xor(stM[s][r], off, 32);
                    float s2 = __shfl_xor(stS[s][r], off, 32);
                    float nm = fmaxf(stM[s][r], m2);
                    stS[s][r] = stS[s][r] * __expf(stM[s][r] - nm) + s2 * __expf(m2 - nm);
                    stM[s][r] = nm;
                }
        }
        if (idx == 0) {
#pragma unroll
            for (int s = 0; s < 2; ++s)
#pragma unroll
                for (int r = 0; r < 8; ++r) {
                    int row = rbase + s * 16 + 8 * half + r;
                    out0[(size_t)row * VS + vs] = stM[s][r];
                    out1[(size_t)row * VS + vs] = stS[s][r];
                }
        }
    } else {
#pragma unroll
        for (int off = 1; off < 16; off <<= 1) {
#pragma unroll
            for (int s = 0; s < 2; ++s)
#pragma unroll
                for (int r = 0; r < 8; ++r)
                    stS[s][r] += __shfl_xor(stS[s][r], off, 32);
        }
        if (idx == 0) {
#pragma unroll
            for (int s = 0; s < 2; ++s)
#pragma unroll
                for (int r = 0; r < 8; ++r) {
                    int row = rbase + s * 16 + 8 * half + r;
                    out0[(size_t)row * VS + vs] = stS[s][r];
                }
        }
    }
}

// combine per-chunk (m,s) partials into final per-row (m,s)
__global__ __launch_bounds__(256) void reduce_ms_kernel(
    const float* __restrict__ pm, const float* __restrict__ ps,
    float* __restrict__ m, float* __restrict__ s)
{
    int i = blockIdx.x * blockDim.x + threadIdx.x;
    if (i >= NROWS) return;
    float mm = -INFINITY, ss = 0.0f;
#pragma unroll
    for (int vs = 0; vs < VS; ++vs) {
        float m2 = pm[(size_t)i * VS + vs];
        float s2 = ps[(size_t)i * VS + vs];
        float nm = fmaxf(mm, m2);
        ss = ss * __expf(mm - nm) + s2 * __expf(m2 - nm);
        mm = nm;
    }
    m[i] = mm;
    s[i] = ss;
}

// p_y[i] = exp(dot(x_i, W[y_i]) + b[y_i] - m_i) / s_i   (one wave per row)
__global__ __launch_bounds__(256) void ylogit_kernel(
    const float* __restrict__ x, const int* __restrict__ y,
    const float* __restrict__ W, const float* __restrict__ bias,
    const float* __restrict__ m, const float* __restrict__ s,
    float* __restrict__ py)
{
    int wave = threadIdx.x >> 5;
    int lane = threadIdx.x & 31;
    int row  = blockIdx.x * 8 + wave;
    if (row >= NROWS) return;
    int yv = y[row];
    const float* xr = x + (size_t)row * EMB;
    const float* wr = W + (size_t)yv * EMB;
    float acc = 0.0f;
#pragma unroll 4
    for (int k = lane; k < EMB; k += 32) acc += xr[k] * wr[k];
#pragma unroll
    for (int off = 16; off >= 1; off >>= 1) acc += __shfl_xor(acc, off, 32);
    if (lane == 0) {
        float l = acc + bias[yv];
        py[row] = __expf(l - m[row]) / s[row];
    }
}

// loss = mean_i( log(sum_vs tp[i][vs]) - py[i] )
__global__ __launch_bounds__(256) void finalize_kernel(
    const float* __restrict__ tp, const float* __restrict__ py,
    float* __restrict__ out)
{
    __shared__ float red[256];
    float acc = 0.0f;
    for (int i = threadIdx.x; i < NROWS; i += 256) {
        float t = 0.0f;
#pragma unroll
        for (int vs = 0; vs < VS; ++vs) t += tp[(size_t)i * VS + vs];
        acc += logf(t) - py[i];
    }
    red[threadIdx.x] = acc;
    __syncthreads();
    for (int o = 128; o > 0; o >>= 1) {
        if (threadIdx.x < o) red[threadIdx.x] += red[threadIdx.x + o];
        __syncthreads();
    }
    if (threadIdx.x == 0) out[0] = red[0] / (float)NROWS;
}

extern "C" void kernel_launch(void* const* d_in, const int* in_sizes, int n_in,
                              void* d_out, int out_size, void* d_ws, size_t ws_size,
                              hipStream_t stream) {
    const float* x  = (const float*)d_in[0];
    const int*   y  = (const int*)d_in[1];
    const float* W  = (const float*)d_in[2];
    const float* b  = (const float*)d_in[3];

    float* wsf = (float*)d_ws;
    float* pm = wsf;                        // NROWS*VS
    float* ps = pm + (size_t)NROWS * VS;    // NROWS*VS
    float* tp = ps + (size_t)NROWS * VS;    // NROWS*VS
    float* m  = tp + (size_t)NROWS * VS;    // NROWS
    float* s  = m + NROWS;                  // NROWS
    float* py = s + NROWS;                  // NROWS

    __bf16* bfbase = (__bf16*)(wsf + NFLOAT);
    __bf16* xh = bfbase;
    __bf16* xl = xh + XN;
    __bf16* wh = xl + XN;
    __bf16* wl = wh + WN;

    dim3 grid(RB, VS);
    const bool pre = (ws_size >= WS_NEED);

    if (pre) {
        split_bf16_kernel<<<(int)((XN / 8 + 255) / 256), 256, 0, stream>>>(x, xh, xl, (long)XN);
        split_bf16_kernel<<<(int)((WN / 8 + 255) / 256), 256, 0, stream>>>(W, wh, wl, (long)WN);
        gemm_softmax_kernel<1, true><<<grid, 256, 0, stream>>>(
            x, W, xh, xl, wh, wl, b, nullptr, nullptr, pm, ps);
        reduce_ms_kernel<<<(NROWS + 255) / 256, 256, 0, stream>>>(pm, ps, m, s);
        gemm_softmax_kernel<2, true><<<grid, 256, 0, stream>>>(
            x, W, xh, xl, wh, wl, b, m, s, tp, nullptr);
    } else {
        gemm_softmax_kernel<1, false><<<grid, 256, 0, stream>>>(
            x, W, nullptr, nullptr, nullptr, nullptr, b, nullptr, nullptr, pm, ps);
        reduce_ms_kernel<<<(NROWS + 255) / 256, 256, 0, stream>>>(pm, ps, m, s);
        gemm_softmax_kernel<2, false><<<grid, 256, 0, stream>>>(
            x, W, nullptr, nullptr, nullptr, nullptr, b, m, s, tp, nullptr);
    }
    ylogit_kernel<<<NROWS / 8, 256, 0, stream>>>(x, y, W, b, m, s, py);
    finalize_kernel<<<1, 256, 0, stream>>>(tp, py, (float*)d_out);
}
